// BSpanDecoder_62354335203756
// MI455X (gfx1250) — matrix operand order repaired and checked by reading, not compiled, not run on hardware
//
#include <hip/hip_runtime.h>
#include <hip/hip_bf16.h>
#include <math.h>

#define B_ 64
#define T_ 512
#define H_ 256
#define E_ 128
#define V_ 32000
#define S_ 128
#define L_ 16
#define FIN_ 640   // 2H + E

typedef __attribute__((ext_vector_type(16))) __bf16 v16bf;
typedef __attribute__((ext_vector_type(8)))  float  v8f;
typedef __attribute__((ext_vector_type(4)))  float  f4v;

// Branch-free tanh: 1 - 2/(exp(2x)+1). v_exp_f32 + v_rcp_f32, saturates to +-1.
__device__ __forceinline__ float fast_tanh(float x) {
  const float e = __expf(2.f * x);
  return 1.f - 2.f * __builtin_amdgcn_rcpf(e + 1.f);
}

// ---- WMMA tile loaders (bf16 16x16x32, wave32, contiguous k order shared by A and B) ----
// A (16x32): lane L holds row (L&15); vector elems i=0..15 -> K = (L>>4)*16 + i.
__device__ __forceinline__ v16bf load_a_tile_f32(const float* __restrict__ rowPtr,
                                                 int k0, int half) {
  v16bf a;
  const float* p = rowPtr + k0 + half * 16;
#pragma unroll
  for (int i = 0; i < 8; ++i) a[i] = (__bf16)p[i];
#pragma unroll
  for (int i = 0; i < 8; ++i) a[8 + i] = (__bf16)p[8 + i];
  return a;
}
// B (32x16): lane L holds column (L&15); vector elems i=0..15 -> K = (L>>4)*16 + i.
__device__ __forceinline__ v16bf load_b_tile_f32(const float* __restrict__ colPtr,
                                                 int k0, int half) {
  v16bf b;
  const float* p = colPtr + k0 + half * 16;
#pragma unroll
  for (int i = 0; i < 16; ++i) b[i] = (__bf16)p[i];
  return b;
}

// ---- Prep: convert W2 = attn_W[:, H:2H] to bf16 once (L2-resident 128KB) ----
__global__ void conv_w2_kernel(const float* __restrict__ attnW, __bf16* __restrict__ w2bf) {
  const int idx = blockIdx.x * 256 + threadIdx.x;   // H_*H_ = 65536 total
  const int n = idx >> 8, k = idx & 255;
  w2bf[idx] = (__bf16)attnW[(size_t)n * (2 * H_) + H_ + k];
}

// ---- Kernel A: hW[b,n] = attn_b[n] + last_hidden[b,:] @ attn_W[n, 0:H] ----
__global__ void hw_kernel(const float* __restrict__ lh, const float* __restrict__ attn_W,
                          const float* __restrict__ attn_b, float* __restrict__ hW) {
  const int b = blockIdx.x, n = threadIdx.x;
  __shared__ float sh[H_];
  sh[n] = lh[b * H_ + n];
  __syncthreads();
  float acc = attn_b[n];
  const float* w = attn_W + (size_t)n * (2 * H_);
  for (int k = 0; k < H_; ++k) acc += sh[k] * w[k];
  hW[b * H_ + n] = acc;
}

// ---- Kernel B: scores[b,t] = sum_n v[n]*tanh(hW[b,n] + enc[b,t,:]@W2[n,:]) ----
// One wave per 16 t-rows of one b.
// Stage the A-strip (16 rows x 256 f32 = 16KB) into LDS via async-to-LDS,
// build bf16 A-fragments ONCE in registers, loop n0 re-loading only bf16 B.
__global__ void attn_score_kernel(const float* __restrict__ u,       // [T,B,H]
                                  const __bf16* __restrict__ w2bf,   // [H,H] bf16
                                  const float* __restrict__ hW,      // [B,H]
                                  const float* __restrict__ attn_v,  // [H]
                                  float* __restrict__ scores)        // [B,T]
{
  const int lane = threadIdx.x & 31;
  const int m    = lane & 15;
  const int half = lane >> 4;
  const int t0   = blockIdx.x * 16;
  const int b    = blockIdx.y;

  __shared__ float sA[16 * H_];   // 16 KB A-strip
  __shared__ float sRow[16];

  // --- async stage: 1024 float4 chunks, 32 lanes x 32 iters, ASYNCcnt-tracked ---
  for (int it = 0; it < 32; ++it) {
    const int f4 = lane + 32 * it;       // float4 index 0..1023
    const int r  = f4 >> 6;              // row (64 float4 per row)
    const int c4 = f4 & 63;
    const float* gp = u + ((size_t)(t0 + r) * B_ + b) * H_ + c4 * 4;
    const unsigned ldsOff = (unsigned)(size_t)(&sA[r * H_ + c4 * 4]); // flat low32 = LDS offset
    asm volatile("global_load_async_to_lds_b128 %0, %1, off"
                 :: "v"(ldsOff), "v"(gp) : "memory");
  }
  asm volatile("s_wait_asynccnt 0" ::: "memory");
  __syncthreads();

  // --- build all 8 bf16 A-fragments once (row m, contiguous K = half*16 + i, same as B) ---
  v16bf afrag[8];
#pragma unroll
  for (int kt = 0; kt < 8; ++kt) {
    const float* pa = &sA[m * H_ + kt * 32 + half * 16];
#pragma unroll
    for (int i = 0; i < 8; ++i) afrag[kt][i] = (__bf16)pa[i];
#pragma unroll
    for (int i = 0; i < 8; ++i) afrag[kt][8 + i] = (__bf16)pa[8 + i];
  }

  const float* hWb = hW + (size_t)b * H_;
  float rs[8];
#pragma unroll
  for (int r = 0; r < 8; ++r) rs[r] = 0.f;

  for (int n0 = 0; n0 < H_; n0 += 16) {
    const __bf16* brow = w2bf + (size_t)(n0 + m) * H_ + half * 16;
    if (n0 + 16 < H_)                           // -> global_prefetch_b8
      __builtin_prefetch(brow + (size_t)16 * H_, 0, 1);
    v8f acc = {};
#pragma unroll
    for (int kt = 0; kt < 8; ++kt) {
      v16bf bv = *(const v16bf*)(brow + kt * 32);   // 32B aligned bf16 fragment
      acc = __builtin_amdgcn_wmma_f32_16x16x32_bf16(false, afrag[kt], false, bv,
                                                    (short)0, acc, false, false);
    }
    const int n    = n0 + m;          // this lane's C-column
    const float vh   = attn_v[n];
    const float bias = hWb[n];
#pragma unroll
    for (int r = 0; r < 8; ++r)
      rs[r] += vh * fast_tanh(acc[r] + bias);   // branch-free, co-executes w/ XDL
  }

  // cross-lane row reduction via LDS float atomics (ds_add_f32)
  if (lane < 16) sRow[lane] = 0.f;
  __syncthreads();
#pragma unroll
  for (int r = 0; r < 8; ++r)
    atomicAdd(&sRow[half * 8 + r], rs[r]);
  __syncthreads();
  if (lane < 16) scores[(size_t)b * T_ + t0 + lane] = sRow[lane];
}

// ---- Kernel C: softmax over T, context, and assemble ffnn_in[:,128:640] ----
__global__ void softmax_ctx_kernel(const float* __restrict__ u,       // [T,B,H]
                                   const float* __restrict__ scores,  // [B,T]
                                   const float* __restrict__ lh,      // [B,H]
                                   float* __restrict__ fin)           // [B,640]
{
  const int b = blockIdx.x, tid = threadIdx.x;  // 256 threads
  __shared__ float red[256];
  __shared__ float wgt[T_];
  const float s0 = scores[(size_t)b * T_ + tid];
  const float s1 = scores[(size_t)b * T_ + 256 + tid];
  red[tid] = fmaxf(s0, s1);
  __syncthreads();
  for (int off = 128; off > 0; off >>= 1) {
    if (tid < off) red[tid] = fmaxf(red[tid], red[tid + off]);
    __syncthreads();
  }
  const float mx = red[0];
  __syncthreads();
  const float e0 = __expf(s0 - mx), e1 = __expf(s1 - mx);
  wgt[tid] = e0; wgt[256 + tid] = e1;
  red[tid] = e0 + e1;
  __syncthreads();
  for (int off = 128; off > 0; off >>= 1) {
    if (tid < off) red[tid] += red[tid + off];
    __syncthreads();
  }
  const float inv = 1.f / red[0];
  float ctx = 0.f;
  for (int t = 0; t < T_; ++t)
    ctx += wgt[t] * u[((size_t)t * B_ + b) * H_ + tid];
  fin[(size_t)b * FIN_ + 128 + tid] = ctx * inv;                 // context slot
  fin[(size_t)b * FIN_ + 384 + tid] = lh[(size_t)b * H_ + tid];  // last_hidden slot
}

// ---- Kernel D: embed_z + ctrl projection -> ffnn_in[:,0:128] ----
__global__ void embed_kernel(const int* __restrict__ z, const float* __restrict__ embW,
                             const float* __restrict__ ctrlW, const float* __restrict__ ctrlb,
                             float* __restrict__ fin) {
  const int b = blockIdx.x, e = threadIdx.x;  // 128 threads
  __shared__ float sh[E_];
  const int tok = z[b];
  sh[e] = embW[(size_t)tok * E_ + e];
  __syncthreads();
  float acc = ctrlb[e];
  const float* w = ctrlW + (size_t)e * E_;
  for (int k = 0; k < E_; ++k) acc += sh[k] * w[k];
  fin[(size_t)b * FIN_ + e] = sh[e] + acc;
}

// ---- Kernel E: hidden = relu(ffnn_in @ Whidᵀ + b)  (WMMA, K=640) ----
__global__ void hidden_gemm_kernel(const float* __restrict__ fin,  // [B,640]
                                   const float* __restrict__ W,    // [H,640]
                                   const float* __restrict__ bias, // [H]
                                   float* __restrict__ hid,        // ws [B,H]
                                   float* __restrict__ dout)       // d_out[0:B*H]
{
  const int lane = threadIdx.x & 31;
  const int m = lane & 15, half = lane >> 4;
  const int m0 = blockIdx.x * 16;   // batch tile
  const int n0 = blockIdx.y * 16;   // hidden-unit tile
  const float* arow = fin + (size_t)(m0 + m) * FIN_;
  const float* brow = W + (size_t)(n0 + m) * FIN_;
  v8f acc = {};
  for (int k0 = 0; k0 < FIN_; k0 += 32) {
    v16bf av = load_a_tile_f32(arow, k0, half);
    v16bf bv = load_b_tile_f32(brow, k0, half);
    acc = __builtin_amdgcn_wmma_f32_16x16x32_bf16(false, av, false, bv,
                                                  (short)0, acc, false, false);
  }
  const int n = n0 + m;
  const float bn = bias[n];
#pragma unroll
  for (int r = 0; r < 8; ++r) {
    const int bb = m0 + half * 8 + r;
    const float v = fmaxf(acc[r] + bn, 0.f);
    hid[(size_t)bb * H_ + n] = v;
    dout[(size_t)bb * H_ + n] = v;   // output #1
  }
}

// ---- Kernel F: gen = relu(hidden @ Woutᵀ + b)  (WMMA, K=256) ----
__global__ void gen_gemm_kernel(const float* __restrict__ hid,  // [B,H]
                                const float* __restrict__ W,    // [S,H]
                                const float* __restrict__ bias, // [S]
                                float* __restrict__ gen)        // [B,S]
{
  const int lane = threadIdx.x & 31;
  const int m = lane & 15, half = lane >> 4;
  const int m0 = blockIdx.x * 16;   // batch tile
  const int n0 = blockIdx.y * 16;   // slot tile
  const float* arow = hid + (size_t)(m0 + m) * H_;
  const float* brow = W + (size_t)(n0 + m) * H_;
  v8f acc = {};
#pragma unroll
  for (int k0 = 0; k0 < H_; k0 += 32) {
    v16bf av = load_a_tile_f32(arow, k0, half);
    v16bf bv = load_b_tile_f32(brow, k0, half);
    acc = __builtin_amdgcn_wmma_f32_16x16x32_bf16(false, av, false, bv,
                                                  (short)0, acc, false, false);
  }
  const int n = n0 + m;
  const float bn = bias[n];
#pragma unroll
  for (int r = 0; r < 8; ++r) {
    const int bb = m0 + half * 8 + r;
    gen[(size_t)bb * S_ + n] = fmaxf(acc[r] + bn, 0.f);
  }
}

// ---- Kernel G: scatter-add + softmax over V, constant base row + NT streaming ----
__global__ void vocab_softmax_kernel(const int* __restrict__ vmap,  // [L,S]
                                     const float* __restrict__ gen, // [B,S]
                                     float* __restrict__ probas)    // [L*B, V]
{
  const int row = blockIdx.x;          // l*B + b
  const int l = row / B_, b = row % B_;
  const int tid = threadIdx.x;         // 256 threads
  __shared__ int   cols[S_];
  __shared__ float g[S_];
  __shared__ float ev[S_];
  __shared__ int   isf[S_];
  __shared__ float redf[256];
  __shared__ int   redi[256];
  if (tid < S_) { cols[tid] = vmap[l * S_ + tid]; g[tid] = gen[(size_t)b * S_ + tid]; }
  __syncthreads();
  float myev = 0.f; int myf = 0;
  if (tid < S_) {
    const int c = cols[tid];
    float acc = -0.01f;
    bool first = true;
    for (int s = 0; s < S_; ++s) {
      if (cols[s] == c) { acc += g[s]; if (s < tid) first = false; }
    }
    if (first) { myev = __expf(acc); myf = 1; }
    ev[tid] = myev; isf[tid] = myf;
  }
  redf[tid] = myev; redi[tid] = myf;
  __syncthreads();
  for (int off = 128; off > 0; off >>= 1) {
    if (tid < off) { redf[tid] += redf[tid + off]; redi[tid] += redi[tid + off]; }
    __syncthreads();
  }
  const float baseE = expf(-0.01f);
  const float denom = (float)(V_ - redi[0]) * baseE + redf[0];
  const float inv   = 1.f / denom;
  const float baseP = baseE * inv;
  float* out = probas + (size_t)row * V_;
  // streaming constant fill: never re-read -> float4 non-temporal stores (512B/instr/wave)
  {
    const f4v basev = {baseP, baseP, baseP, baseP};
    f4v* out4 = (f4v*)out;
    for (int v4 = tid; v4 < V_ / 4; v4 += 256)
      __builtin_nontemporal_store(basev, &out4[v4]);
  }
  __syncthreads();
  if (tid < S_ && isf[tid]) out[cols[tid]] = ev[tid] * inv;  // patch specials
}

extern "C" void kernel_launch(void* const* d_in, const int* in_sizes, int n_in,
                              void* d_out, int out_size, void* d_ws, size_t ws_size,
                              hipStream_t stream) {
  const float* u     = (const float*)d_in[0];   // u_enc_out [T,B,H]
  const float* lh    = (const float*)d_in[1];   // last_hidden [1,B,H]
  const int*   z     = (const int*)d_in[2];     // z_tm1 [1,B]
  const int*   vmap  = (const int*)d_in[3];     // slot_vocab_map [L,S]
  const float* embW  = (const float*)d_in[4];   // [V,E]
  const float* ctrlW = (const float*)d_in[5];   // [E,E]
  const float* ctrlb = (const float*)d_in[6];   // [E]
  const float* attnW = (const float*)d_in[7];   // [H,2H]
  const float* attnb = (const float*)d_in[8];   // [H]
  const float* attnv = (const float*)d_in[9];   // [H]
  const float* hidW  = (const float*)d_in[10];  // [H,640]
  const float* hidb  = (const float*)d_in[11];  // [H]
  const float* outW  = (const float*)d_in[12];  // [S,H]
  const float* outb  = (const float*)d_in[13];  // [S]
  (void)in_sizes; (void)n_in; (void)out_size; (void)ws_size;

  float* ws     = (float*)d_ws;
  float* hW     = ws;                  // B*H    = 16384 floats
  float* scores = hW + B_ * H_;        // B*T    = 32768
  float* fin    = scores + B_ * T_;    // B*640  = 40960
  float* hid    = fin + B_ * FIN_;     // B*H    = 16384
  float* gen    = hid + B_ * H_;       // B*S    = 8192
  __bf16* w2bf  = (__bf16*)(gen + B_ * S_);  // H*H bf16 = 128KB (32B-aligned)

  float* outHidden = (float*)d_out;                 // [B,H]
  float* outProbas = outHidden + B_ * H_;           // [L*B, V]

  hipLaunchKernelGGL(conv_w2_kernel,      dim3(H_*H_/256), dim3(256), 0, stream, attnW, w2bf);
  hipLaunchKernelGGL(hw_kernel,           dim3(B_),        dim3(H_),  0, stream, lh, attnW, attnb, hW);
  hipLaunchKernelGGL(attn_score_kernel,   dim3(T_/16, B_), dim3(32),  0, stream, u, w2bf, hW, attnv, scores);
  hipLaunchKernelGGL(softmax_ctx_kernel,  dim3(B_),        dim3(256), 0, stream, u, scores, lh, fin);
  hipLaunchKernelGGL(embed_kernel,        dim3(B_),        dim3(E_),  0, stream, z, embW, ctrlW, ctrlb, fin);
  hipLaunchKernelGGL(hidden_gemm_kernel,  dim3(4, 16),     dim3(32),  0, stream, fin, hidW, hidb, hid, outHidden);
  hipLaunchKernelGGL(gen_gemm_kernel,     dim3(4, 8),      dim3(32),  0, stream, hid, outW, outb, gen);
  hipLaunchKernelGGL(vocab_softmax_kernel,dim3(L_ * B_),   dim3(256), 0, stream, vmap, gen, outProbas);
}
